// MultiHeadAttention_89232240541956
// MI455X (gfx1250) — matrix operand order repaired and checked by reading, not compiled, not run on hardware
//
#include <hip/hip_runtime.h>

// ---------------------------------------------------------------------------
// Problem constants (from reference): B=8, S=4096, H=1024, LOCAL=GLOBAL=64,
// ATTN=1024, EXT=1152. Output = scores [B,S] fp32. v-branch is dead code.
// ---------------------------------------------------------------------------
#define SEQ      4096
#define HDIM     1024
#define EXTD     1152
#define ATTND    1024
#define MT       128                 // token rows per workgroup
#define ASTRIDE  1160                // padded bf16 row stride in LDS (2320 B, != 0 mod 64 banks)
#define KT       36                  // EXT / 32 k-steps
#define NT16     64                  // ATTN / 16 n-tiles
#define PACK_PER (NT16 * KT * 32)    // fragments-lanes per matrix = 73728

typedef __bf16        v16bf __attribute__((ext_vector_type(16)));
typedef float         v8f   __attribute__((ext_vector_type(8)));
typedef unsigned int  v4u   __attribute__((ext_vector_type(4)));
typedef unsigned int  v2u   __attribute__((ext_vector_type(2)));
typedef float         v4f   __attribute__((ext_vector_type(4)));

union Frag { v4u u[2]; v16bf v; };   // 32 bytes = 16 bf16 = one WMMA A/B operand

__device__ __forceinline__ unsigned short f2bf(float f) {
  unsigned u = __builtin_bit_cast(unsigned, f);
  u += 0x7FFFu + ((u >> 16) & 1u);   // round-to-nearest-even
  return (unsigned short)(u >> 16);
}

__device__ __forceinline__ v2u pack4(v4f f) {
  v2u r;
  r.x = (unsigned)f2bf(f.x) | ((unsigned)f2bf(f.y) << 16);
  r.y = (unsigned)f2bf(f.z) | ((unsigned)f2bf(f.w) << 16);
  return r;
}

// ---------------------------------------------------------------------------
// Pre-pass: fp32 W[EXT][ATTN] -> bf16, pre-swizzled into WMMA B-fragment
// layout: [tile_n(64)][tile_k(36)][lane(32)][16 bf16].
//   lane 0..15 : N = tn*16+lane,     K = tk*32 + 0..15
//   lane 16..31: N = tn*16+lane-16,  K = tk*32 + 16..31
// ---------------------------------------------------------------------------
__global__ __launch_bounds__(256) void pack_w_kernel(
    const float* __restrict__ Wq, const float* __restrict__ Wk,
    unsigned short* __restrict__ outQ, unsigned short* __restrict__ outK)
{
  int tid = blockIdx.x * 256 + threadIdx.x;
  if (tid >= 2 * PACK_PER) return;
  const float* W      = (tid < PACK_PER) ? Wq : Wk;
  unsigned short* o   = (tid < PACK_PER) ? outQ : outK;
  int t   = (tid < PACK_PER) ? tid : tid - PACK_PER;
  int tn  = t / (KT * 32);
  int r   = t - tn * (KT * 32);
  int tk  = r >> 5;
  int ln  = r & 31;
  int n   = tn * 16 + (ln & 15);
  int kb  = tk * 32 + ((ln >> 4) << 4);
  alignas(16) unsigned short tmp[16];
#pragma unroll
  for (int e = 0; e < 16; ++e)
    tmp[e] = f2bf(W[(size_t)(kb + e) * ATTND + n]);
  v4u* dst = (v4u*)(o + (size_t)t * 16);
  dst[0] = *(const v4u*)&tmp[0];
  dst[1] = *(const v4u*)&tmp[8];
}

// ---------------------------------------------------------------------------
// Fused kernel: build bf16 x_ext tile in LDS (290KB, resident), GEMM both
// Wq and Wk with v_wmma_f32_16x16x32_bf16, fuse bias+relu+diag(qk^T)+mask.
// Block = 256 threads = 8 waves; wave w owns rows [16w, 16w+16).
// ---------------------------------------------------------------------------
__global__ __launch_bounds__(256) void fused_qk_scores(
    const float* __restrict__ h, const int* __restrict__ mask,
    const float* __restrict__ g, const float* __restrict__ l,
    const float* __restrict__ bq, const float* __restrict__ bk,
    const unsigned short* __restrict__ wqP, const unsigned short* __restrict__ wkP,
    float* __restrict__ out)
{
  extern __shared__ unsigned short sA[];          // [MT][ASTRIDE] bf16
  const int tid  = threadIdx.x;
  const int t0   = blockIdx.x * MT;               // first token of this tile
  const int bIdx = t0 >> 12;                      // batch (128 | 4096 => uniform)

  // ---- stage x_ext tile into LDS as bf16 (h | g | l) ----
  const v4f* h4 = (const v4f*)h;
#pragma unroll 4
  for (int idx = tid; idx < MT * 256; idx += 256) {          // h: 1024 cols
    int row = idx >> 8, c4 = idx & 255;
    v4f f = h4[(size_t)(t0 + row) * 256 + c4];
    *(v2u*)(&sA[row * ASTRIDE + c4 * 4]) = pack4(f);
  }
  const v4f* g4 = (const v4f*)g;
  for (int idx = tid; idx < MT * 16; idx += 256) {           // g: cols 1024..1087
    int row = idx >> 4, c4 = idx & 15;
    v4f f = g4[bIdx * 16 + c4];
    *(v2u*)(&sA[row * ASTRIDE + 1024 + c4 * 4]) = pack4(f);
  }
  const v4f* l4 = (const v4f*)l;
  for (int idx = tid; idx < MT * 16; idx += 256) {           // l: cols 1088..1151
    int row = idx >> 4, c4 = idx & 15;
    v4f f = l4[(size_t)(t0 + row) * 16 + c4];
    *(v2u*)(&sA[row * ASTRIDE + 1088 + c4 * 4]) = pack4(f);
  }
  __syncthreads();

  const int wave = tid >> 5;
  const int lane = tid & 31;
  const int mrow = wave * 16;
  // A-fragment base for this lane (contiguous k order, same as packed W):
  //   lanes 0-15: row=mrow+lane,    K = kc+0..15
  //   lanes16-31: row=mrow+lane-16, K = kc+16..31
  const unsigned short* aBase =
      &sA[(mrow + (lane & 15)) * ASTRIDE + ((lane >> 4) << 4)];

  float rowAcc[8];
#pragma unroll
  for (int i = 0; i < 8; ++i) rowAcc[i] = 0.f;

  for (int n0 = 0; n0 < ATTND; n0 += 64) {                   // 16 N-chunks
    v8f accQ[4], accK[4];
#pragma unroll
    for (int j = 0; j < 4; ++j) { accQ[j] = 0; accK[j] = 0; }

    for (int kc = 0; kc < EXTD; kc += 32) {                  // 36 K-steps
      Frag a;
      a.u[0] = *(const v4u*)(aBase + kc);                    // ds_load_b128
      a.u[1] = *(const v4u*)(aBase + kc + 8);                // ds_load_b128
      const int tk = kc >> 5;
#pragma unroll
      for (int j = 0; j < 4; ++j) {
        const size_t fo = ((size_t)(((n0 >> 4) + j) * KT + tk) * 32 + lane) * 16;
        Frag fb;
        const v4u* pq = (const v4u*)(wqP + fo);
        fb.u[0] = pq[0]; fb.u[1] = pq[1];
        __builtin_prefetch((const void*)(wqP + fo + 512), 0, 0);  // next k-step
        accQ[j] = __builtin_amdgcn_wmma_f32_16x16x32_bf16(
            false, a.v, false, fb.v, (short)0, accQ[j], false, false);
        const v4u* pk = (const v4u*)(wkP + fo);
        fb.u[0] = pk[0]; fb.u[1] = pk[1];
        __builtin_prefetch((const void*)(wkP + fo + 512), 0, 0);
        accK[j] = __builtin_amdgcn_wmma_f32_16x16x32_bf16(
            false, a.v, false, fb.v, (short)0, accK[j], false, false);
      }
    }

    // epilogue for this 64-wide chunk: bias + relu + elementwise q*k
    // C/D layout: VGPR i, lanes 0-15 -> (M=i, N=lane); lanes 16-31 -> (M=i+8)
#pragma unroll
    for (int j = 0; j < 4; ++j) {
      const int n = n0 + j * 16 + (lane & 15);
      const float bqv = bq[n];
      const float bkv = bk[n];
#pragma unroll
      for (int i = 0; i < 8; ++i) {
        float qv = fmaxf(accQ[j][i] + bqv, 0.f);
        float kv = fmaxf(accK[j][i] + bkv, 0.f);
        rowAcc[i] += qv * kv;
      }
    }
  }

  // reduce the per-lane partial dot products across the 16 lanes sharing a row
#pragma unroll
  for (int i = 0; i < 8; ++i) {
    float v = rowAcc[i];
    v += __shfl_xor(v, 1);
    v += __shfl_xor(v, 2);
    v += __shfl_xor(v, 4);
    v += __shfl_xor(v, 8);
    rowAcc[i] = v;
  }

  if ((lane & 15) == 0) {
    const int half = lane >> 4;                              // 0: rows 0-7, 1: rows 8-15
#pragma unroll
    for (int i = 0; i < 8; ++i) {
      const int t  = t0 + mrow + half * 8 + i;
      float sc = rowAcc[i] * 0.03125f;                       // 1/sqrt(1024)
      out[t] = (mask[t] == 1) ? -1e9f : sc;
    }
  }
}

// ---------------------------------------------------------------------------
extern "C" void kernel_launch(void* const* d_in, const int* in_sizes, int n_in,
                              void* d_out, int out_size, void* d_ws, size_t ws_size,
                              hipStream_t stream) {
  const float* h    = (const float*)d_in[0];
  const int*   mask = (const int*)  d_in[1];
  const float* g    = (const float*)d_in[2];
  const float* l    = (const float*)d_in[3];
  const float* Wq   = (const float*)d_in[4];
  const float* bq   = (const float*)d_in[5];
  const float* Wk   = (const float*)d_in[6];
  const float* bk   = (const float*)d_in[7];
  // d_in[8]=Wv, d_in[9]=bv are dead code in the reference's return path.
  float* out = (float*)d_out;

  unsigned short* wqP = (unsigned short*)d_ws;                       // 2.36 MB
  unsigned short* wkP = wqP + (size_t)PACK_PER * 16;                 // 2.36 MB

  pack_w_kernel<<<(2 * PACK_PER + 255) / 256, 256, 0, stream>>>(Wq, Wk, wqP, wkP);

  const size_t smem = (size_t)MT * ASTRIDE * sizeof(unsigned short); // 296,960 B < 320 KB
  const int nblocks = (8 * SEQ) / MT;                                // 256 workgroups
  fused_qk_scores<<<nblocks, 256, smem, stream>>>(h, mask, g, l, bq, bk, wqP, wkP, out);
}
